// RudyWithMacros_13030930776416
// MI455X (gfx1250) — matrix operand order repaired and checked by reading, not compiled, not run on hardware
//
#include <hip/hip_runtime.h>

#define NBX 512
#define NBY 512
#define NUM_NETS 131072
#define NUM_MOVABLE 50000
#define NUM_TERMINALS 5000
#define NUM_NODES (NUM_MOVABLE + NUM_TERMINALS)
#define NUM_PINS (NUM_NETS * 4)
#define NUM_MACROS 96
#define K_TOTAL (NUM_NETS + NUM_MACROS)   // 131168 = 4099 * 32 exactly
#define NCHUNK (K_TOTAL / 32)             // 4099
#define KSPLIT 16
#define TILE_X 128                         // 8 waves x 16 rows
#define TILE_Y 64                          // 4 N-subtiles
#define ROUTING_HW 50000.0f

typedef __attribute__((ext_vector_type(16))) __bf16    v16bf;
typedef __attribute__((ext_vector_type(2)))  __bf16    bf16x2;
typedef __attribute__((ext_vector_type(8)))  float     v8f;
typedef __attribute__((ext_vector_type(8)))  unsigned  v8u;

// ---------------- zero scratch ----------------
__global__ void k_zero(float* p, int n) {
  int i = blockIdx.x * blockDim.x + threadIdx.x;
  if (i < n) p[i] = 0.0f;
}

// ---------------- per-net / per-macro bounding boxes -> compact descriptors ----------------
// netdata[n*6 + {0..5}] = { xmin*512, xmax*512, ymin*512, ymax*512, wh, wv }
__global__ void k_bbox(const float* __restrict__ pin_pos,
                       const float* __restrict__ net_weights,
                       const int*   __restrict__ flat_netpin,
                       const float* __restrict__ pos,
                       const float* __restrict__ nsx,
                       const float* __restrict__ nsy,
                       const int*   __restrict__ macro_idx,
                       float* __restrict__ netdata) {
  int n = blockIdx.x * blockDim.x + threadIdx.x;
  if (n >= K_TOTAL) return;
  float xmin, xmax, ymin, ymax, wh, wv;
  if (n < NUM_NETS) {
    float w = net_weights[n];
    xmin = 1e30f; xmax = -1e30f; ymin = 1e30f; ymax = -1e30f;
    #pragma unroll
    for (int i = 0; i < 4; ++i) {
      int p = flat_netpin[n * 4 + i];
      float x = pin_pos[p];
      float y = pin_pos[NUM_PINS + p];
      xmin = fminf(xmin, x); xmax = fmaxf(xmax, x);
      ymin = fminf(ymin, y); ymax = fmaxf(ymax, y);
    }
    wh = w / (ymax - ymin);
    wv = w / (xmax - xmin);
  } else {
    int m = macro_idx[n - NUM_NETS];
    float x = pos[m], y = pos[NUM_NODES + m];
    float sx = nsx[m], sy = nsy[m];
    xmin = x; xmax = x + sx; ymin = y; ymax = y + sy;
    float inva = 1.0f / (sx * sy);
    wh = 10.0f * inva;   // MACRO_UTIL_H / area
    wv = 10.0f * inva;   // MACRO_UTIL_V / area
  }
  float* o = netdata + n * 6;
  o[0] = xmin * (float)NBX; o[1] = xmax * (float)NBX;
  o[2] = ymin * (float)NBY; o[3] = ymax * (float)NBY;
  o[4] = wh;                o[5] = wv;
}

__device__ __forceinline__ unsigned pack_bf16(float a, float b) {
  bf16x2 p = { (__bf16)a, (__bf16)b };     // build_vector(fptrunc,fptrunc) -> v_cvt_pk_bf16_f32
  return __builtin_bit_cast(unsigned, p);
}

// ---------------- WMMA rasterization GEMM: H/V[x,y] = sum_k A[k,x]*B[k,y] ----------------
// 256 threads = 8 waves. Wave w owns x-strip [bx*128 + w*16, +16) x [by*64, +64).
// B tile (32k x 64y) generated ONCE per workgroup, in fragment order, in LDS.
// 4x8 output tiles * KSPLIT k-partials; partials combined via f32 atomics.
__global__ __launch_bounds__(256)
void k_rudy_wmma(const float* __restrict__ netdata,
                 float* __restrict__ Hacc,
                 float* __restrict__ Vacc) {
  __shared__ float snet[32 * 6];
  __shared__ __align__(32) unsigned Blds[4][32][8];  // [y-subtile][lane][j-pair], 4 KB
  const int tid  = threadIdx.x;
  const int lane = tid & 31;
  const int wave = tid >> 5;
  const int lm   = lane & 15;
  const int hi   = lane >> 4;

  const int X0 = blockIdx.x * TILE_X + wave * 16;
  const int Y0 = blockIdx.y * TILE_Y;
  const float xf = (float)(X0 + lm);

  v8f cH[4] = {};
  v8f cV[4] = {};

  const int per = (NCHUNK + KSPLIT - 1) / KSPLIT;
  int c0 = (int)blockIdx.z * per;
  int c1 = c0 + per; if (c1 > NCHUNK) c1 = NCHUNK;

  for (int c = c0; c < c1; ++c) {
    __syncthreads();                       // previous iteration done with snet/Blds
    for (int i = tid; i < 192; i += 256) snet[i] = netdata[c * 192 + i];
    if (tid == 0 && (c + 1) < c1) __builtin_prefetch(&netdata[(c + 1) * 192], 0, 1);
    __syncthreads();                       // snet ready

    // ---- cooperative B tile in fragment order: element (t,ln,j) -> k=(ln>>4)*16+j, y=Y0+t*16+(ln&15)
    #pragma unroll
    for (int p = 0; p < 4; ++p) {
      int slot    = tid + p * 256;         // 0..1023
      int pairIdx = slot & 7;
      int ln      = (slot >> 3) & 31;
      int t       = slot >> 8;
      int kb      = ((ln >> 4) << 4) | (pairIdx * 2);
      float yf = (float)(Y0 + t * 16 + (ln & 15));
      float o0 = fmaxf(fminf(snet[kb * 6 + 3], yf + 1.0f) - fmaxf(snet[kb * 6 + 2], yf), 0.0f);
      float o1 = fmaxf(fminf(snet[kb * 6 + 9], yf + 1.0f) - fmaxf(snet[kb * 6 + 8], yf), 0.0f);
      Blds[t][ln][pairIdx] = pack_bf16(o0, o1);
    }

    // ---- per-wave A fragments (16-bit A 16x32): element j -> K = (hi<<4) | j
    float fH[16], fV[16];
    #pragma unroll
    for (int j = 0; j < 16; ++j) {
      int k = (hi << 4) | j;
      float ox = fmaxf(fminf(snet[k * 6 + 1], xf + 1.0f) - fmaxf(snet[k * 6 + 0], xf), 0.0f);
      fH[j] = ox * snet[k * 6 + 4];
      fV[j] = ox * snet[k * 6 + 5];
    }
    v8u auH, auV;
    #pragma unroll
    for (int j2 = 0; j2 < 8; ++j2) {
      auH[j2] = pack_bf16(fH[2 * j2], fH[2 * j2 + 1]);
      auV[j2] = pack_bf16(fV[2 * j2], fV[2 * j2 + 1]);
    }
    v16bf aH = __builtin_bit_cast(v16bf, auH);
    v16bf aV = __builtin_bit_cast(v16bf, auV);

    __syncthreads();                       // Blds ready

    #pragma unroll
    for (int t = 0; t < 4; ++t) {
      v16bf bF = *(const v16bf*)&Blds[t][lane][0];   // 2x ds_load_b128
      cH[t] = __builtin_amdgcn_wmma_f32_16x16x32_bf16(false, aH, false, bF,
                                                      (short)0, cH[t], false, false);
      cV[t] = __builtin_amdgcn_wmma_f32_16x16x32_bf16(false, aV, false, bF,
                                                      (short)0, cV[t], false, false);
    }
  }

  // C/D layout: VGPR r -> row M = 8*hi + r; lane lm -> column N
  #pragma unroll
  for (int t = 0; t < 4; ++t) {
    int y = Y0 + t * 16 + lm;
    #pragma unroll
    for (int r = 0; r < 8; ++r) {
      int x = X0 + hi * 8 + r;
      atomicAdd(&Hacc[x * NBY + y], cH[t][r]);
      atomicAdd(&Vacc[x * NBY + y], cV[t][r]);
    }
  }
}

// ---------------- blur along x (dim 0), reflect padding, normalization folded in ----------------
__global__ void k_blur_x(const float* __restrict__ in, float* __restrict__ out, float scale) {
  int idx = blockIdx.x * blockDim.x + threadIdx.x;
  if (idx >= NBX * NBY) return;
  int x = idx / NBY, y = idx - x * NBY;
  float w = expf(-0.5f / (32.0f * 32.0f));      // sigma = 512/16 = 32
  float norm = 1.0f / (1.0f + 2.0f * w);
  int xm = (x == 0)       ? 1       : x - 1;
  int xp = (x == NBX - 1) ? NBX - 2 : x + 1;
  out[idx] = (w * in[xm * NBY + y] + in[idx] + w * in[xp * NBY + y]) * norm * scale;
}

// ---------------- blur along y + route map + overflow bin counting ----------------
__global__ void k_finalize(const float* __restrict__ tH, const float* __restrict__ tV,
                           float* __restrict__ out, int* __restrict__ cnts) {
  __shared__ int sh[2];
  int tid = threadIdx.x;
  if (tid < 2) sh[tid] = 0;
  __syncthreads();
  int idx = blockIdx.x * blockDim.x + tid;
  if (idx < NBX * NBY) {
    int x = idx / NBY, y = idx - x * NBY;
    float w = expf(-0.5f / (32.0f * 32.0f));
    float norm = 1.0f / (1.0f + 2.0f * w);
    int ym = (y == 0)       ? 1       : y - 1;
    int yp = (y == NBY - 1) ? NBY - 2 : y + 1;
    float Hb = (w * tH[x * NBY + ym] + tH[idx] + w * tH[x * NBY + yp]) * norm;
    float Vb = (w * tV[x * NBY + ym] + tV[idx] + w * tV[x * NBY + yp]) * norm;
    out[idx] = fmaxf(fabsf(Hb), fabsf(Vb));
    if (Hb > 1.0f) atomicAdd(&sh[0], 1);
    if (Vb > 1.0f) atomicAdd(&sh[1], 1);
  }
  __syncthreads();
  if (tid == 0 && sh[0]) atomicAdd(&cnts[0], sh[0]);
  if (tid == 1 && sh[1]) atomicAdd(&cnts[1], sh[1]);
}

__global__ void k_scalars(const int* __restrict__ cnts, float* __restrict__ out) {
  int h = cnts[0], v = cnts[1];
  out[NBX * NBY + 0] = (float)(h > v ? h : v);   // max_overflow
  out[NBX * NBY + 1] = (float)(h + v);           // total_overflow
}

extern "C" void kernel_launch(void* const* d_in, const int* in_sizes, int n_in,
                              void* d_out, int out_size, void* d_ws, size_t ws_size,
                              hipStream_t stream) {
  const float* pos     = (const float*)d_in[0];
  const float* pin_pos = (const float*)d_in[1];
  const float* nsx     = (const float*)d_in[2];
  const float* nsy     = (const float*)d_in[3];
  const float* nw      = (const float*)d_in[4];
  const int*   fnp     = (const int*)  d_in[5];
  const int*   mi      = (const int*)  d_in[6];
  float* out = (float*)d_out;

  float* netdata = (float*)d_ws;              // K_TOTAL*6 floats (~3.1 MB)
  float* Hacc = netdata + (size_t)K_TOTAL * 6;
  float* Vacc = Hacc + NBX * NBY;
  float* tH   = Vacc + NBX * NBY;
  float* tV   = tH   + NBX * NBY;
  int*   cnts = (int*)(tV + NBX * NBY);

  // zero accumulators + counters (bitwise zero int == 0.0f)
  int nz = 4 * NBX * NBY + 2;
  k_zero<<<(nz + 255) / 256, 256, 0, stream>>>(Hacc, nz);

  k_bbox<<<(K_TOTAL + 255) / 256, 256, 0, stream>>>(
      pin_pos, nw, fnp, pos, nsx, nsy, mi, netdata);

  dim3 grid(NBX / TILE_X, NBY / TILE_Y, KSPLIT);  // 4 x 8 tiles x 16-way K split = 512 WGs
  k_rudy_wmma<<<grid, 256, 0, stream>>>(netdata, Hacc, Vacc);

  // 512*512 normalization cancels the 512-scaling of ox and oy: H_norm = Hacc / ROUTING
  int nel = NBX * NBY;
  k_blur_x<<<(nel + 255) / 256, 256, 0, stream>>>(Hacc, tH, 1.0f / ROUTING_HW);
  k_blur_x<<<(nel + 255) / 256, 256, 0, stream>>>(Vacc, tV, 1.0f / ROUTING_HW);

  k_finalize<<<(nel + 255) / 256, 256, 0, stream>>>(tH, tV, out, cnts);
  k_scalars<<<1, 1, 0, stream>>>(cnts, out);
}
